// GrammarFieldOperator_47132971107201
// MI455X (gfx1250) — compile-verified, matrix operand order checked by reading, not run on hardware
//
#include <hip/hip_runtime.h>

// ---------------------------------------------------------------------------
// MoE "grammar field" operator for MI455X (gfx1250, wave32, WMMA + TDM).
// out = x + sum_e softmax(x@gate_w)[.,e] * (x @ W_e + b_e)
// Main GEMM: bf16 v_wmma_f32_16x16x32_bf16, double-buffered LDS tiles fed by
// the Tensor Data Mover (tensor_load_to_lds, TENSORcnt) when available.
// ---------------------------------------------------------------------------

typedef __bf16 bf16;
typedef __attribute__((ext_vector_type(16))) __bf16 v16bf;
typedef __attribute__((ext_vector_type(8)))  __bf16 v8bf;
typedef __attribute__((ext_vector_type(8)))  float  v8f;
typedef __attribute__((ext_vector_type(4)))  unsigned int v4u;
typedef __attribute__((ext_vector_type(8)))  int v8i;
typedef __attribute__((ext_vector_type(4)))  int v4i;

union FragBF16 { v16bf v; v8bf h[2]; };

#define NTOK 8192     // B*L
#define DD   1024     // d_model
#define NE   16       // experts
#define BM   128      // rows per block tile
#define BN   128      // cols per block tile
#define BK   32       // K chunk (== one wmma K step for bf16)
#define KPAD 40       // BK + 8 bf16 pad -> 80B row stride, conflict-free b128
#define NCHUNK (NE * (DD / BK))   // 512 pipeline steps

#if __has_builtin(__builtin_amdgcn_tensor_load_to_lds) && \
    __has_builtin(__builtin_amdgcn_s_wait_tensorcnt)
#define USE_TDM 1
#else
#define USE_TDM 0
#endif

__device__ __forceinline__ v8f vzero8() {
  v8f v;
#pragma unroll
  for (int k = 0; k < 8; ++k) v[k] = 0.0f;
  return v;
}

// ---- kernel 1: fp32 -> bf16 convert of x ---------------------------------
__global__ void k_convert_x(const float* __restrict__ x, bf16* __restrict__ xb, int n) {
  int i = blockIdx.x * blockDim.x + threadIdx.x;
  int stride = gridDim.x * blockDim.x;
  for (; i < n; i += stride) xb[i] = (bf16)x[i];
}

// ---- kernel 2: experts_w [K=1024][F=16384] fp32 -> wT [F][K] bf16 --------
__global__ void k_transpose_w(const float* __restrict__ w, bf16* __restrict__ wT) {
  __shared__ float tile[32][33];
  const int f0 = blockIdx.x * 32;
  const int k0 = blockIdx.y * 32;
  const int tx = threadIdx.x & 31;
  const int ty = threadIdx.x >> 5;            // 0..7
#pragma unroll
  for (int r = 0; r < 4; ++r) {
    int kr = ty + 8 * r;
    tile[kr][tx] = w[(k0 + kr) * (NE * DD) + f0 + tx];
  }
  __syncthreads();
#pragma unroll
  for (int r = 0; r < 4; ++r) {
    int fr = ty + 8 * r;
    wT[(f0 + fr) * DD + k0 + tx] = (bf16)tile[tx][fr];
  }
}

// ---- kernel 3: gate logits + softmax over 16 experts ---------------------
__global__ void k_gate(const float* __restrict__ x, const float* __restrict__ gw,
                       float* __restrict__ wgt) {
  int gid = blockIdx.x * blockDim.x + threadIdx.x;   // < NTOK*NE
  int n = gid >> 4;
  int e = gid & 15;
  const float* xr = x + n * DD;
  float acc = 0.0f;
#pragma unroll 8
  for (int k = 0; k < DD; ++k) acc = fmaf(xr[k], gw[k * NE + e], acc);
  float m = acc;
#pragma unroll
  for (int off = 8; off; off >>= 1) m = fmaxf(m, __shfl_xor(m, off, 16));
  float p = __expf(acc - m);
  float s = p;
#pragma unroll
  for (int off = 8; off; off >>= 1) s += __shfl_xor(s, off, 16);
  wgt[gid] = p / s;
}

#if USE_TDM
// Issue one 2D TDM tile load: tile_d0 x tile_d1 bf16 elements, contiguous in
// dim0, row stride `stride0` elements; LDS rows padded 64B data + 16B pad
// (pad_interval = 16 dwords, pad_amount = 4 dwords) -> KPAD*2 = 80B stride.
__device__ __forceinline__ void tdm_load_2d(unsigned lds_off,
                                            unsigned long long gaddr,
                                            unsigned tensor_d0, unsigned tensor_d1,
                                            unsigned stride0,
                                            unsigned tile_d0, unsigned tile_d1) {
  v4u g0;
  g0[0] = 1u;                                            // count=1, user mode
  g0[1] = lds_off;                                       // lds_addr (bytes)
  g0[2] = (unsigned)(gaddr & 0xffffffffull);             // global_addr[31:0]
  g0[3] = (unsigned)((gaddr >> 32) & 0x01ffffffull)      // global_addr[56:32]
        | (2u << 30);                                    // type = 2 (image)
  v8i g1;
  g1[0] = (int)((1u << 16)      // data_size = 2 bytes (bf16)
              | (1u << 20)      // pad_enable
              | (3u << 22)      // pad_interval: 16 dwords (64B)
              | (3u << 25));    // pad_amount:   4 dwords (16B)
  g1[1] = (int)((tensor_d0 & 0xffffu) << 16);                    // dim0 lo
  g1[2] = (int)((tensor_d0 >> 16) | ((tensor_d1 & 0xffffu) << 16));
  g1[3] = (int)((tensor_d1 >> 16) | (tile_d0 << 16));
  g1[4] = (int)(tile_d1 & 0xffffu);                              // tile_dim2=0
  g1[5] = (int)stride0;                                          // dim0 stride
  g1[6] = 0;
  g1[7] = 0;
  v4i gz = {0, 0, 0, 0};
#if defined(__clang_major__) && __clang_major__ >= 23
  v8i gz8 = {0, 0, 0, 0, 0, 0, 0, 0};
  __builtin_amdgcn_tensor_load_to_lds(g0, g1, gz, gz, gz8, 0);
#else
  __builtin_amdgcn_tensor_load_to_lds(g0, g1, gz, gz, 0);
#endif
}
#endif

// ---- kernel 4: main MoE GEMM + weighted combine + bias + residual --------
// 256 threads = 8 waves; tile 128x128; wave: 32 rows x 64 cols = 2x4 frags.
// Pipeline: 512 (expert,k-chunk) steps, double-buffered LDS, TDM prefetch of
// chunk ch+1 overlapped with WMMA compute of chunk ch.
__global__ __launch_bounds__(256) void k_moe(
    const bf16* __restrict__ xb,   // [NTOK][DD] bf16
    const bf16* __restrict__ wT,   // [NE*DD][DD] bf16 (expert cols x K)
    const float* __restrict__ wgt, // [NTOK][NE] softmax weights
    const float* __restrict__ eb,  // [NE*DD] bias
    const float* __restrict__ x,   // [NTOK][DD] fp32 (residual)
    float* __restrict__ out)       // [NTOK][DD] fp32
{
  __shared__ bf16  sA[2][BM * KPAD];
  __shared__ bf16  sB[2][BN * KPAD];
  __shared__ float sW[NE * BM];

  const int tid  = threadIdx.x;
  const int lane = tid & 31;
  const int wid  = tid >> 5;
  const int row0 = blockIdx.y * BM;
  const int col0 = blockIdx.x * BN;
  const int m_off = (wid & 3) * 32;
  const int n_off = (wid >> 2) * 64;

  // stage gate weights, transposed: sW[e*BM + r] (unrolled: no exec-loop)
#pragma unroll
  for (int t = 0; t < 8; ++t) {
    int i = tid + t * 256;
    int e = i >> 7, r = i & 127;
    sW[e * BM + r] = wgt[(row0 + r) * NE + e];
  }

  const int lm  = lane & 15;           // fragment row/col within 16
  const int hi8 = (lane >> 4) << 3;    // 0 for lanes 0-15, 8 for 16-31

  v8f sum[2][4];
#pragma unroll
  for (int i = 0; i < 2; ++i)
#pragma unroll
    for (int j = 0; j < 4; ++j) sum[i][j] = vzero8();

  v8f c[2][4];

#if USE_TDM
  const unsigned long long xbase = (unsigned long long)(uintptr_t)xb;
  const unsigned long long wbase = (unsigned long long)(uintptr_t)wT;
  const unsigned ldsA[2] = {(unsigned)(uintptr_t)&sA[0][0],
                            (unsigned)(uintptr_t)&sA[1][0]};
  const unsigned ldsB[2] = {(unsigned)(uintptr_t)&sB[0][0],
                            (unsigned)(uintptr_t)&sB[1][0]};
  if (wid == 0) {  // prologue: chunk 0 -> buffer 0
    tdm_load_2d(ldsA[0], xbase + (unsigned long long)(row0 * DD) * 2,
                DD, NTOK, DD, BK, BM);
    tdm_load_2d(ldsB[0], wbase + (unsigned long long)(col0 * DD) * 2,
                DD, NE * DD, DD, BK, BN);
  }
#endif

  for (int ch = 0; ch < NCHUNK; ++ch) {
    const int e  = ch >> 5;
    const int kc = ch & 31;
    const int p  = ch & 1;

    if (kc == 0) {
#pragma unroll
      for (int i = 0; i < 2; ++i)
#pragma unroll
        for (int j = 0; j < 4; ++j) c[i][j] = vzero8();
    }

    __syncthreads();  // all waves done with buffer 1-p -> free to overwrite

#if USE_TDM
    if (wid == 0) {
      if (ch + 1 < NCHUNK) {
        const int en = (ch + 1) >> 5, kn = (ch + 1) & 31, pn = (ch + 1) & 1;
        tdm_load_2d(ldsA[pn],
                    xbase + ((unsigned long long)row0 * DD + kn * BK) * 2,
                    DD, NTOK, DD, BK, BM);
        tdm_load_2d(ldsB[pn],
                    wbase + (((unsigned long long)(en * DD + col0)) * DD + kn * BK) * 2,
                    DD, NE * DD, DD, BK, BN);
        __builtin_amdgcn_s_wait_tensorcnt(2);  // chunk ch landed (in-order)
      } else {
        __builtin_amdgcn_s_wait_tensorcnt(0);
      }
    }
#else
    {  // fallback: register-staged copy, all loads in flight before stores
      const int r0 = tid >> 2, s0 = (tid & 3) << 3;  // 16B segment
      const int r1 = r0 + 64;
      v8bf a0 = *(const v8bf*)&xb[(row0 + r0) * DD + kc * BK + s0];
      v8bf a1 = *(const v8bf*)&xb[(row0 + r1) * DD + kc * BK + s0];
      v8bf b0 = *(const v8bf*)&wT[(e * DD + col0 + r0) * DD + kc * BK + s0];
      v8bf b1 = *(const v8bf*)&wT[(e * DD + col0 + r1) * DD + kc * BK + s0];
      *(v8bf*)&sA[p][r0 * KPAD + s0] = a0;
      *(v8bf*)&sA[p][r1 * KPAD + s0] = a1;
      *(v8bf*)&sB[p][r0 * KPAD + s0] = b0;
      *(v8bf*)&sB[p][r1 * KPAD + s0] = b1;
    }
#endif

    __syncthreads();  // publish chunk ch to all waves

    // fragment loads: lane L holds row lm, K = hi8..hi8+7 and hi8+16..+23
    FragBF16 a[2], b[4];
#pragma unroll
    for (int i = 0; i < 2; ++i) {
      const bf16* pa = &sA[p][(m_off + 16 * i + lm) * KPAD + hi8];
      a[i].h[0] = *(const v8bf*)pa;
      a[i].h[1] = *(const v8bf*)(pa + 16);
    }
#pragma unroll
    for (int j = 0; j < 4; ++j) {
      const bf16* pb = &sB[p][(n_off + 16 * j + lm) * KPAD + hi8];
      b[j].h[0] = *(const v8bf*)pb;
      b[j].h[1] = *(const v8bf*)(pb + 16);
    }
#pragma unroll
    for (int i = 0; i < 2; ++i)
#pragma unroll
      for (int j = 0; j < 4; ++j)
        c[i][j] = __builtin_amdgcn_wmma_f32_16x16x32_bf16(
            false, a[i].v, false, b[j].v, (short)0, c[i][j], false, false);

    if (kc == 31) {
      // scale expert's C by per-row gate weight, fold into sum.
      // C layout: VGPR k <-> row (m_off + 16i + hi8 + k).
#pragma unroll
      for (int i = 0; i < 2; ++i) {
        const float* pw = &sW[e * BM + m_off + 16 * i + hi8];
        v8f wv;
#pragma unroll
        for (int k = 0; k < 8; ++k) wv[k] = pw[k];
#pragma unroll
        for (int j = 0; j < 4; ++j) sum[i][j] += wv * c[i][j];
      }
    }
  }

  // epilogue: bias (sum_e w_e * b_e[col]) + residual + store
#pragma unroll
  for (int j = 0; j < 4; ++j) {
    const int cidx = col0 + n_off + 16 * j + lm;
    float bcol[NE];
#pragma unroll
    for (int e = 0; e < NE; ++e) bcol[e] = eb[e * DD + cidx];
#pragma unroll
    for (int i = 0; i < 2; ++i) {
#pragma unroll
      for (int k = 0; k < 8; ++k) {
        const int rl = m_off + 16 * i + hi8 + k;
        float bias = 0.0f;
#pragma unroll
        for (int e = 0; e < NE; ++e) bias = fmaf(sW[e * BM + rl], bcol[e], bias);
        const int g = (row0 + rl) * DD + cidx;
        out[g] = x[g] + sum[i][j][k] + bias;
      }
    }
  }
}

// ---------------------------------------------------------------------------
extern "C" void kernel_launch(void* const* d_in, const int* in_sizes, int n_in,
                              void* d_out, int out_size, void* d_ws, size_t ws_size,
                              hipStream_t stream) {
  const float* x   = (const float*)d_in[0];   // [4,2048,1024]
  const float* gw  = (const float*)d_in[1];   // [1024,16]
  const float* ew  = (const float*)d_in[2];   // [1024,16384]
  const float* eb  = (const float*)d_in[3];   // [16384]
  float* out = (float*)d_out;

  const size_t XB_BYTES = (size_t)NTOK * DD * sizeof(bf16);      // 16 MB
  const size_t WT_BYTES = (size_t)NE * DD * DD * sizeof(bf16);   // 32 MB
  bf16*  xb  = (bf16*)d_ws;
  bf16*  wT  = (bf16*)((char*)d_ws + XB_BYTES);
  float* wgt = (float*)((char*)d_ws + XB_BYTES + WT_BYTES);      // 512 KB

  const int n_x = NTOK * DD;
  k_convert_x<<<4096, 256, 0, stream>>>(x, xb, n_x);
  k_transpose_w<<<dim3((NE * DD) / 32, DD / 32), 256, 0, stream>>>(ew, wT);
  k_gate<<<(NTOK * NE) / 256, 256, 0, stream>>>(x, gw, wgt);
  k_moe<<<dim3(DD / BN, NTOK / BM), 256, 0, stream>>>(xb, wT, wgt, eb, x, out);
}
